// Contrastive_38079180046371
// MI455X (gfx1250) — matrix operand order repaired and checked by reading, not compiled, not run on hardware
//
#include <hip/hip_runtime.h>

// Contrastive loss pipeline for MI455X (gfx1250, wave32).
// Dominant einsum runs on v_wmma_f32_16x16x32_bf16 (bf16 in, f32 accum),
// B panel staged to LDS via global_load_async_to_lds_b128 when available.
// Workspace requirement: ~85.5 MB.

#define EPSF    1e-8f
#define MARGINF 0.2f

#if defined(__gfx1250__) && __has_builtin(__builtin_amdgcn_global_load_async_to_lds_b128)
#define HAVE_ASYNC_LDS 1
#else
#define HAVE_ASYNC_LDS 0
#endif

typedef __attribute__((ext_vector_type(16))) __bf16 v16bf;
typedef __attribute__((ext_vector_type(8)))  float  v8f;
typedef int v4i_ __attribute__((vector_size(16)));   // matches builtin param type

__device__ __forceinline__ unsigned short f2bf(float f) {
  union { float f; unsigned u; } x; x.f = f;
  unsigned r = x.u + 0x7FFFu + ((x.u >> 16) & 1u);   // round-to-nearest-even
  return (unsigned short)(r >> 16);
}

__device__ __forceinline__ float warp_sum(float v) {
#pragma unroll
  for (int o = 16; o > 0; o >>= 1) v += __shfl_xor(v, o, 32);
  return v;
}

__device__ __forceinline__ float block_sum(float v, float* red) {
  int t = threadIdx.x;
  red[t] = v; __syncthreads();
  for (int s = blockDim.x >> 1; s > 0; s >>= 1) {
    if (t < s) red[t] += red[t + s];
    __syncthreads();
  }
  float r = red[0]; __syncthreads();
  return r;
}

__device__ __forceinline__ float block_max(float v, float* red) {
  int t = threadIdx.x;
  red[t] = v; __syncthreads();
  for (int s = blockDim.x >> 1; s > 0; s >>= 1) {
    if (t < s) red[t] = fmaxf(red[t], red[t + s]);
    __syncthreads();
  }
  float r = red[0]; __syncthreads();
  return r;
}

// ---------------- sentences: center, norm, normalize, bf16 copy ------------
__global__ __launch_bounds__(256) void k_prep_sent(
    const float* __restrict__ sent, float* __restrict__ S, float* __restrict__ Sn,
    unsigned short* __restrict__ Snbf, float* __restrict__ normS) {
  __shared__ float red[256];
  int b = blockIdx.x, t = threadIdx.x;
  float s = 0.f;
  for (int c = t; c < 512; c += 256) s += sent[b * 512 + c];
  float mean = block_sum(s, red) * (1.f / 512.f);
  float q = 0.f;
  for (int c = t; c < 512; c += 256) {
    float x = sent[b * 512 + c] - mean;
    S[b * 512 + c] = x;
    q += x * x;
  }
  float nrm = sqrtf(block_sum(q, red));
  float rn = 1.f / fmaxf(nrm, EPSF);
  for (int c = t; c < 512; c += 256) {
    float xn = (sent[b * 512 + c] - mean) * rn;
    Sn[b * 512 + c] = xn;
    Snbf[b * 512 + c] = f2bf(xn);
  }
  if (t == 0) normS[b] = nrm;
}

// ---- videos: per (b,p) mean/norm over c; write Vn as bf16 in [b][p][c] ----
__global__ __launch_bounds__(256) void k_prep_vid(
    const float* __restrict__ videos, float* __restrict__ mu,
    float* __restrict__ vnorm, unsigned int* __restrict__ Vnbf) {
  int b = blockIdx.y;
  int p = blockIdx.x * 8 + (threadIdx.x >> 5);
  int lane = threadIdx.x & 31;
  const float* base = videos + (size_t)b * 512 * 1024 + p; // [b][c][p]
  float s = 0.f;
  for (int c = lane; c < 512; c += 32) s += base[(size_t)c * 1024];
  s = warp_sum(s);
  float m = s * (1.f / 512.f);
  float q = 0.f;
  for (int c = lane; c < 512; c += 32) {
    float x = base[(size_t)c * 1024] - m;
    q += x * x;
  }
  q = warp_sum(q);
  float nrm = sqrtf(q);
  float rn = 1.f / fmaxf(nrm, EPSF);
  if (lane == 0) { mu[b * 1024 + p] = m; vnorm[b * 1024 + p] = nrm; }
  unsigned int* outp = Vnbf + ((size_t)(b * 1024 + p)) * 256; // 256 dwords = 512 bf16
  for (int cp = lane; cp < 256; cp += 32) {
    float x0 = (base[(size_t)(2 * cp) * 1024] - m) * rn;
    float x1 = (base[(size_t)(2 * cp + 1) * 1024] - m) * rn;
    outp[cp] = (unsigned)f2bf(x0) | ((unsigned)f2bf(x1) << 16);
  }
}

// -------- cross[i,j,p] = Sn(64x512) @ Vn[j](512x1024) via bf16 WMMA --------
// grid = (64 p-tiles, 64 j), block = 128 (4 waves); wave w -> rows i0=16w.
// B panel (16 p-rows x K=512 bf16 = 16KB, contiguous in [j][p][c]) is staged
// to LDS once per block (async-to-LDS when available) and shared by 4 waves.
__global__ __launch_bounds__(128) void k_gemm_cross(
    const unsigned int* __restrict__ Snbf,  // [64][256] dword pairs of bf16
    const unsigned int* __restrict__ Vnbf,  // [64][1024][256]
    const float* __restrict__ mask, float* __restrict__ cross) {
  __shared__ unsigned ldsB[16 * 260];  // 16 rows x 256 dwords, +4 pad (bank spread)
  const int lane = threadIdx.x & 31;
  const int wave = threadIdx.x >> 5;
  const int j  = blockIdx.y;
  const int p0 = blockIdx.x << 4;
  const int i0 = wave << 4;
  const int hf  = lane >> 4;
  const int l15 = lane & 15;

  // ---- stage B panel: global -> LDS ----
  const unsigned int* gsrc = Vnbf + ((size_t)j * 1024 + p0) * 256;
  {
    const int r   = threadIdx.x >> 3;  // 16 rows, 8 threads per row
    const int ch0 = threadIdx.x & 7;   // 64 x 16B chunks per row
#if HAVE_ASYNC_LDS
    for (int ch = ch0; ch < 64; ch += 8) {
      __builtin_amdgcn_global_load_async_to_lds_b128(
          (v4i_*)(gsrc + r * 256 + ch * 4),
          (v4i_*)(ldsB + r * 260 + ch * 4),
          0, 0);
    }
#if __has_builtin(__builtin_amdgcn_s_wait_asynccnt)
    __builtin_amdgcn_s_wait_asynccnt(0);
#else
    asm volatile("s_wait_asynccnt 0" ::: "memory");
#endif
#else
    for (int ch = ch0; ch < 64; ch += 8) {
#pragma unroll
      for (int d = 0; d < 4; ++d)
        ldsB[r * 260 + ch * 4 + d] = gsrc[r * 256 + ch * 4 + d];
    }
#endif
  }
  __syncthreads();

  v8f acc = {};
  const unsigned int* arow = Snbf + (size_t)(i0 + l15) * 256;  // A: M = l15

  for (int c0 = 0; c0 < 512; c0 += 32) {
    const int cp = c0 >> 1;
    union { v16bf v; unsigned int u[8]; } A, Bm;
#pragma unroll
    for (int vv = 0; vv < 8; ++vv) {
      // A 16x32 layout (matches B): K = 2v + 16*half
      int ka = vv + 8 * hf;
      A.u[vv] = arow[cp + ka];
    }
    // B 32x16 layout (contiguous K, same as A): K = 2v + 16*half -> dwords [cp+8*hf .. +7]
    const int fb = l15 * 260 + cp + 8 * hf;
#pragma unroll
    for (int vv = 0; vv < 8; ++vv) Bm.u[vv] = ldsB[fb + vv];

    acc = __builtin_amdgcn_wmma_f32_16x16x32_bf16(
        /*neg_a=*/false, A.v, /*neg_b=*/false, Bm.v,
        /*c_mod=*/(short)0, acc, /*reuse_a=*/false, /*reuse_b=*/false);
    if (c0 + 32 < 512)
      __builtin_prefetch(arow + ((c0 + 32) >> 1), 0, 1); // global_prefetch_b8
  }
  // D layout: N = lane&15, M = e + 8*(lane>=16)
  const int n = p0 + l15;
  const float mval = mask[n];
#pragma unroll
  for (int e = 0; e < 8; ++e) {
    int m = i0 + e + 8 * hf;
    float val = acc[e];
    if (mval == 0.f) val = -__builtin_inff();
    cross[((size_t)m * 64 + j) * 1024 + n] = val;
  }
}

// --- self_score -> pmap + softmax w; video_pos[b,c] = sum_p w*Vf[b,c,p] ----
__global__ __launch_bounds__(256) void k_vpos(
    const float* __restrict__ videos, const float* __restrict__ cross,
    const float* __restrict__ mu, float* __restrict__ out,
    float* __restrict__ video_pos, float* __restrict__ vpnorm,
    float* __restrict__ vpsum) {
  __shared__ float sw[1024];
  __shared__ float red[256];
  __shared__ float wpart[16];
  int b = blockIdx.x, t = threadIdx.x;
  const float* selfrow = cross + ((size_t)b * 64 + b) * 1024;
  float mx = -__builtin_inff();
  for (int p = t; p < 1024; p += 256) {
    float v = selfrow[p];
    out[1 + b * 1024 + p] = v;          // pmap output
    sw[p] = v;
    mx = fmaxf(mx, v);
  }
  mx = block_max(mx, red);
  float es = 0.f;
  for (int p = t; p < 1024; p += 256) {
    float e = expf(sw[p] - mx);
    sw[p] = e; es += e;
  }
  float tot = block_sum(es, red);
  float rt = 1.f / tot;
  for (int p = t; p < 1024; p += 256) sw[p] *= rt;
  __syncthreads();
  float wm = 0.f;
  for (int p = t; p < 1024; p += 256) wm += sw[p] * mu[b * 1024 + p];
  float wdotmu = block_sum(wm, red);
  int wave = t >> 5, lane = t & 31;
  float s1 = 0.f, s2 = 0.f;
  for (int c = wave; c < 512; c += 8) {
    const float* row = videos + ((size_t)(b * 512 + c)) * 1024;
    float a = 0.f;
    for (int p = lane; p < 1024; p += 32) a += sw[p] * row[p];
    a = warp_sum(a);
    if (lane == 0) {
      float vp = a - wdotmu;            // Vf = videos - mu, folded
      video_pos[b * 512 + c] = vp;
      s1 += vp; s2 += vp * vp;
    }
  }
  if (lane == 0) { wpart[wave] = s1; wpart[8 + wave] = s2; }
  __syncthreads();
  if (t == 0) {
    float a1 = 0.f, a2 = 0.f;
    for (int k = 0; k < 8; ++k) { a1 += wpart[k]; a2 += wpart[8 + k]; }
    vpsum[b] = a1;
    vpnorm[b] = sqrtf(a2);
  }
}

// ------------------- sim = Sn @ Sn^T, per-row argsort ----------------------
__global__ __launch_bounds__(256) void k_sim_sort(
    const float* __restrict__ Sn, int* __restrict__ idxs,
    int* __restrict__ nn_idx, int* __restrict__ gate) {
  __shared__ float sim[64 * 64];
  int t = threadIdx.x;
  for (int e = t; e < 4096; e += 256) {
    int i = e >> 6, j = e & 63;
    const float* a = Sn + i * 512;
    const float* b2 = Sn + j * 512;
    float d = 0.f;
    for (int c = 0; c < 512; ++c) d += a[c] * b2[c];
    sim[e] = d;
  }
  __syncthreads();
  if (t < 64) {
    float v[64]; int id[64];
    for (int j = 0; j < 64; ++j) { v[j] = sim[t * 64 + j]; id[j] = j; }
    for (int r = 0; r < 64; ++r) {           // selection sort, descending
      int best = r;
      for (int q = r + 1; q < 64; ++q) if (v[q] > v[best]) best = q;
      float tv = v[r]; v[r] = v[best]; v[best] = tv;
      int ti = id[r]; id[r] = id[best]; id[best] = ti;
      idxs[t * 64 + r] = id[r];
    }
    nn_idx[t] = id[1];
    gate[t] = (v[1] > 0.9f) ? 1 : 0;
  }
}

// ----------------------------- vas hinge loss ------------------------------
__global__ __launch_bounds__(256) void k_vas(
    const float* __restrict__ video_pos, const float* __restrict__ S,
    const float* __restrict__ normS, const float* __restrict__ vpnorm,
    const int* __restrict__ idxs, float* __restrict__ vas_part) {
  __shared__ float red[256];
  __shared__ float posS;
  int b = blockIdx.x, t = threadIdx.x;
  float d = 0.f;
  for (int c = t; c < 512; c += 256) d += video_pos[b * 512 + c] * S[b * 512 + c];
  float dot = block_sum(d, red);
  if (t == 0) posS = dot / (fmaxf(vpnorm[b], EPSF) * fmaxf(normS[b], EPSF));
  __syncthreads();
  int wave = t >> 5, lane = t & 31;
  float contrib = 0.f;
  for (int k = wave; k < 53; k += 8) {      // idx[:, 6:59] -> 53 negatives
    int x = idxs[b * 64 + 6 + k];
    float dn = 0.f;
    for (int c = lane; c < 512; c += 32) dn += video_pos[b * 512 + c] * S[x * 512 + c];
    dn = warp_sum(dn);
    if (lane == 0) {
      float neg = dn / (fmaxf(vpnorm[b], EPSF) * fmaxf(normS[x], EPSF));
      contrib += fmaxf(0.f, MARGINF + neg - posS);
    }
  }
  float tot = block_sum((lane == 0) ? contrib : 0.f, red);
  if (t == 0) vas_part[b] = tot * (1.f / 53.f);
}

// ------------------------------- crov loss ---------------------------------
__global__ __launch_bounds__(256) void k_crov(
    const float* __restrict__ videos, const float* __restrict__ video_pos,
    const float* __restrict__ mu, const float* __restrict__ vnorm,
    const float* __restrict__ vpnorm, const float* __restrict__ vpsum,
    const int* __restrict__ nn_idx, const int* __restrict__ gate,
    const float* __restrict__ mask, const int* __restrict__ valid_num,
    float* __restrict__ crov_part) {
  __shared__ float cs[1024];
  __shared__ float mk[1024];
  __shared__ float red[256];
  __shared__ float wp[32];
  int b = blockIdx.x, t = threadIdx.x;
  if (!gate[b]) { if (t == 0) crov_part[b] = 0.f; return; }
  int nn = nn_idx[b];
  int wave = t >> 5, lane = t & 31;
  const float* vrow = videos + (size_t)nn * 512 * 1024;
  float vpn = fmaxf(vpnorm[b], EPSF);
  float vsum = vpsum[b];
  for (int p = t; p < 1024; p += 256) mk[p] = mask[p];
  // pass 1: cos(video_pos, pse[:,p]) per p
  for (int p = wave; p < 1024; p += 8) {
    float dsum = 0.f;
    for (int c = lane; c < 512; c += 32)
      dsum += video_pos[b * 512 + c] * vrow[(size_t)c * 1024 + p];
    dsum = warp_sum(dsum);
    if (lane == 0) {
      float dot = dsum - mu[nn * 1024 + p] * vsum;  // pse = vrow - mu folded
      cs[p] = dot / (vpn * fmaxf(vnorm[nn * 1024 + p], EPSF));
    }
  }
  __syncthreads();
  // softmax over p -> vsim (in cs)
  float mx = -__builtin_inff();
  for (int p = t; p < 1024; p += 256) mx = fmaxf(mx, cs[p]);
  mx = block_max(mx, red);
  float es = 0.f;
  for (int p = t; p < 1024; p += 256) { float e = expf(cs[p] - mx); cs[p] = e; es += e; }
  float tot = block_sum(es, red);
  float rt = 1.f / tot;
  for (int p = t; p < 1024; p += 256) cs[p] *= rt;
  __syncthreads();
  float a1 = 0.f, a2 = 0.f;
  for (int p = t; p < 1024; p += 256) {
    float vs = cs[p], m = mu[nn * 1024 + p];
    a1 += vs * m;
    a2 += (1.f - vs) * mk[p] * m;
  }
  float S1 = block_sum(a1, red);
  float S2 = block_sum(a2, red);
  float rinv = 1.f / (float)(*valid_num - 1);
  // pass 2: p_v / n_v dot & norm accumulation
  float dp = 0.f, pp = 0.f, dn2 = 0.f, nq = 0.f;
  for (int c = wave; c < 512; c += 8) {
    const float* row = vrow + (size_t)c * 1024;
    float q1 = 0.f, q2 = 0.f;
    for (int p = lane; p < 1024; p += 32) {
      float vx = row[p], vs = cs[p];
      q1 += vs * vx;
      q2 += (1.f - vs) * mk[p] * vx;
    }
    q1 = warp_sum(q1); q2 = warp_sum(q2);
    if (lane == 0) {
      float pv = q1 - S1;
      float nv = (q2 - S2) * rinv;
      float vc = video_pos[b * 512 + c];
      dp += vc * pv; pp += pv * pv;
      dn2 += vc * nv; nq += nv * nv;
    }
  }
  if (lane == 0) { wp[wave] = dp; wp[8 + wave] = pp; wp[16 + wave] = dn2; wp[24 + wave] = nq; }
  __syncthreads();
  if (t == 0) {
    float Dp = 0.f, Pp = 0.f, Dn = 0.f, Nq = 0.f;
    for (int k = 0; k < 8; ++k) { Dp += wp[k]; Pp += wp[8 + k]; Dn += wp[16 + k]; Nq += wp[24 + k]; }
    float cos_p = Dp / (vpn * fmaxf(sqrtf(Pp), EPSF));
    float cos_n = Dn / (vpn * fmaxf(sqrtf(Nq), EPSF));
    crov_part[b] = fmaxf(0.f, MARGINF + cos_n - cos_p);
  }
}

// ---------- per (i,j): sorted top-128, decay product, nms score ------------
__global__ __launch_bounds__(256) void k_topk(
    const float* __restrict__ cross, const float* __restrict__ iou,
    const float* __restrict__ lam_p, float* __restrict__ scores) {
  __shared__ float val[1024];
  __shared__ int   ind[1024];
  __shared__ float red[256];
  __shared__ float wk[128];
  __shared__ float dec[128];
  int i = blockIdx.y, j = blockIdx.x, t = threadIdx.x;
  const float* src = cross + ((size_t)i * 64 + j) * 1024;
  for (int p = t; p < 1024; p += 256) { val[p] = src[p]; ind[p] = p; }
  __syncthreads();
  // bitonic sort, descending by value (carrying indices)
  for (int k = 2; k <= 1024; k <<= 1) {
    for (int s = k >> 1; s > 0; s >>= 1) {
      for (int q = t; q < 1024; q += 256) {
        int x = q ^ s;
        if (x > q) {
          float a = val[q], c = val[x];
          bool up = ((q & k) == 0);
          if (up ? (a < c) : (a > c)) {
            val[q] = c; val[x] = a;
            int ti = ind[q]; ind[q] = ind[x]; ind[x] = ti;
          }
        }
      }
      __syncthreads();
    }
  }
  float lam = lam_p[0];
  float e = 0.f;
  if (t < 128) { e = expf(lam * (val[t] - val[0])); wk[t] = e; }
  float tot = block_sum(e, red);
  if (t < 128) {
    wk[t] /= tot;
    float pr = 1.f;
    const float* irow = iou + (size_t)ind[t] * 1024;
    for (int kk = 0; kk < t; ++kk) pr *= (1.f - irow[ind[kk]]); // tril product
    dec[t] = pr;
  }
  __syncthreads();
  float part = (t < 128) ? wk[t] * dec[t] * val[t] : 0.f;
  float sc = block_sum(part, red);
  if (t == 0) scores[i * 64 + j] = sc;
}

// ------------------ final contrastive reduction + loss ---------------------
__global__ __launch_bounds__(64) void k_final(
    const float* __restrict__ scores, const float* __restrict__ vas_part,
    const float* __restrict__ crov_part, float* __restrict__ out) {
  __shared__ float r1[64], r2[64], r3[64], r4[64];
  int t = threadIdx.x;
  float diag_t = scores[t * 64 + t];
  float mr = 0.f, mc = 0.f;
  for (int j = 0; j < 64; ++j) {
    if (j == t) continue;
    mr = fmaxf(mr, fmaxf(0.f, MARGINF + scores[t * 64 + j] - diag_t)); // cost_s row
    mc = fmaxf(mc, fmaxf(0.f, MARGINF + scores[j * 64 + t] - diag_t)); // cost_im col
  }
  r1[t] = mr; r2[t] = mc; r3[t] = vas_part[t]; r4[t] = crov_part[t];
  __syncthreads();
  if (t == 0) {
    float s = 0.f, va = 0.f, cr = 0.f;
    for (int i = 0; i < 64; ++i) { s += r1[i] + r2[i]; va += r3[i]; cr += r4[i]; }
    out[0] = s * (1.f / 64.f) + va * (1.f / 64.f) + cr * (1.f / 64.f);
  }
}

extern "C" void kernel_launch(void* const* d_in, const int* in_sizes, int n_in,
                              void* d_out, int out_size, void* d_ws, size_t ws_size,
                              hipStream_t stream) {
  (void)in_sizes; (void)n_in; (void)out_size; (void)ws_size;
  const float* videos    = (const float*)d_in[0];
  const float* sentences = (const float*)d_in[1];
  const float* lam       = (const float*)d_in[2];
  const float* mask      = (const float*)d_in[3];
  const int*   valid_num = (const int*)d_in[4];
  const float* iou       = (const float*)d_in[5];
  float* out = (float*)d_out;

  char* w = (char*)d_ws;
  float* mu        = (float*)w; w += (size_t)64 * 1024 * 4;
  float* vnorm     = (float*)w; w += (size_t)64 * 1024 * 4;
  float* S         = (float*)w; w += (size_t)64 * 512 * 4;
  float* Sn        = (float*)w; w += (size_t)64 * 512 * 4;
  float* normS     = (float*)w; w += 256;
  unsigned short* Snbf = (unsigned short*)w; w += (size_t)64 * 512 * 2;
  unsigned int*   Vnbf = (unsigned int*)w;   w += (size_t)64 * 1024 * 512 * 2;
  float* cross     = (float*)w; w += (size_t)64 * 64 * 1024 * 4;
  float* video_pos = (float*)w; w += (size_t)64 * 512 * 4;
  float* vpnorm    = (float*)w; w += 256;
  float* vpsum     = (float*)w; w += 256;
  int*   idxs      = (int*)w;   w += (size_t)64 * 64 * 4;
  int*   nn_idx    = (int*)w;   w += 256;
  int*   gate      = (int*)w;   w += 256;
  float* scores    = (float*)w; w += (size_t)64 * 64 * 4;
  float* vas_part  = (float*)w; w += 256;
  float* crov_part = (float*)w; w += 256;

  k_prep_sent<<<64, 256, 0, stream>>>(sentences, S, Sn, Snbf, normS);
  k_prep_vid<<<dim3(128, 64), 256, 0, stream>>>(videos, mu, vnorm, Vnbf);
  k_gemm_cross<<<dim3(64, 64), 128, 0, stream>>>((const unsigned int*)Snbf, Vnbf, mask, cross);
  k_vpos<<<64, 256, 0, stream>>>(videos, cross, mu, out, video_pos, vpnorm, vpsum);
  k_sim_sort<<<1, 256, 0, stream>>>(Sn, idxs, nn_idx, gate);
  k_vas<<<64, 256, 0, stream>>>(video_pos, S, normS, vpnorm, idxs, vas_part);
  k_crov<<<64, 256, 0, stream>>>(videos, video_pos, mu, vnorm, vpnorm, vpsum,
                                 nn_idx, gate, mask, valid_num, crov_part);
  k_topk<<<dim3(64, 64), 256, 0, stream>>>(cross, iou, lam, scores);
  k_final<<<1, 64, 0, stream>>>(scores, vas_part, crov_part, out);
}
